// KmeansAttention_44710609551422
// MI455X (gfx1250) — matrix operand order repaired and checked by reading, not compiled, not run on hardware
//
#include <hip/hip_runtime.h>
#include <hip/hip_bf16.h>
#include <math.h>

// ---- problem constants (fixed by the harness/reference) ----
#define BB   4
#define HH   16
#define TT   8192
#define DD   64
#define CC   64       // clusters
#define WSZ  128      // window size
#define NCW  64       // windows per (b,h) == TT/WSZ == CC

typedef __attribute__((ext_vector_type(16))) __bf16 v16bf;
typedef __attribute__((ext_vector_type(8)))  float  v8f;
typedef __attribute__((ext_vector_type(2)))  float  v2f;
typedef unsigned int u32x4 __attribute__((ext_vector_type(4)));
typedef int i32x8 __attribute__((ext_vector_type(8)));
typedef int i32x4 __attribute__((ext_vector_type(4)));

union BF16x16 { unsigned short u[16]; v16bf v; };

__device__ __forceinline__ unsigned short f2bf(float f) {
    unsigned int u = __float_as_uint(f);
    unsigned int r = u + 0x7FFFu + ((u >> 16) & 1u);   // round-to-nearest-even
    return (unsigned short)(r >> 16);
}

// bf16 A fragment: 16x32 tile, row = (lane&15):
//   e=0..7  -> K = kb*32 + half*8 + e ;  e=8..15 -> K = kb*32 + 16 + half*8 + (e-8)
__device__ __forceinline__ v16bf frag_a(const unsigned short* row, int kb, int half) {
    BF16x16 f;
    const unsigned short* p = row + kb * 32 + half * 8;
#pragma unroll
    for (int i = 0; i < 8; ++i) { f.u[i] = p[i]; f.u[8 + i] = p[16 + i]; }
    return f.v;
}

// bf16 B fragment: 32x16 tile, N = (lane&15) selects source row:
//   e=0..7  -> K = kb*32 + half*8 + e ;  e=8..15 -> K = kb*32 + 16 + half*8 + (e-8)
__device__ __forceinline__ v16bf frag_b(const unsigned short* row, int kb, int half) {
    BF16x16 f;
    const unsigned short* p = row + kb * 32 + half * 8;
#pragma unroll
    for (int i = 0; i < 8; ++i) { f.u[i] = p[i]; f.u[8 + i] = p[16 + i]; }
    return f.v;
}

// ------------------------------------------------------------------
// shared helper: f32 WMMA sim tile compute (M=256 tokens, N=64, K=64)
//   sKn: 256 x KS f32 (normalized tokens), sMn: 64 x KS f32 (means rows)
//   sSim: 256 x KS f32 output
// A frag (16x4 f32): lane(l15) = M row, element e: K = kb*4 + half*2 + e
// B frag (4x16 f32): lane(l15) = N row, element e: K = kb*4 + half*2 + e
// ------------------------------------------------------------------
#define KS 68

__device__ __forceinline__ void sim_wmma(const float* sKn, const float* sMn,
                                         float* sSim, int tid) {
    const int lane = tid & 31, wv = tid >> 5;
    const int half = lane >> 4, l15 = lane & 15;
    for (int mt = wv; mt < 16; mt += 8) {
        const float* arow = sKn + (mt * 16 + l15) * KS + half * 2;
        for (int nt = 0; nt < 4; ++nt) {
            const float* brow = sMn + (nt * 16 + l15) * KS + half * 2;
            v8f acc = {0.f, 0.f, 0.f, 0.f, 0.f, 0.f, 0.f, 0.f};
#pragma unroll
            for (int kb = 0; kb < 16; ++kb) {
                v2f a, b2;
                a[0] = arow[kb * 4]; a[1] = arow[kb * 4 + 1];
                b2[0] = brow[kb * 4]; b2[1] = brow[kb * 4 + 1];
                acc = __builtin_amdgcn_wmma_f32_16x16x4_f32(false, a, false, b2,
                                                            (short)0, acc, false, false);
            }
#pragma unroll
            for (int r = 0; r < 8; ++r)
                sSim[(mt * 16 + half * 8 + r) * KS + nt * 16 + l15] = acc[r];
        }
    }
}

// ------------------------------------------------------------------
// K1: per-token l2norm + argmax cluster (f32 WMMA), accumulate bins/sums
// ------------------------------------------------------------------
__global__ __launch_bounds__(256) void kmr_assign(const float* __restrict__ qk,
                                                  const float* __restrict__ means,
                                                  float* __restrict__ g_sums,
                                                  int* __restrict__ g_bins) {
    extern __shared__ char smA[];
    float* sKn   = (float*)smA;                 // 256 x KS
    float* sMn   = sKn + 256 * KS;              // 64 x KS
    float* sSim  = sMn + 64 * KS;               // 256 x KS
    float* sSums = sSim + 256 * KS;             // 64 x 64
    int*   sBins = (int*)(sSums + 64 * 64);     // 64

    const int tid = threadIdx.x;
    const long gid0 = (long)blockIdx.x * 256;
    const int h_idx = (int)((gid0 / TT) % HH);

    for (int i = tid; i < CC * DD; i += 256) {
        sMn[(i >> 6) * KS + (i & 63)] = means[(long)h_idx * CC * DD + i];
        sSums[i] = 0.f;
    }
    if (tid < CC) sBins[tid] = 0;

    const long gid = gid0 + tid;
    const float4* qp = (const float4*)(qk + gid * DD);
    float4 qv[16];
    float ss = 0.f;
#pragma unroll
    for (int i = 0; i < 16; ++i) {
        qv[i] = qp[i];
        ss += qv[i].x * qv[i].x + qv[i].y * qv[i].y + qv[i].z * qv[i].z + qv[i].w * qv[i].w;
    }
    const float inv = 1.0f / fmaxf(sqrtf(ss), 1e-12f);
#pragma unroll
    for (int i = 0; i < 16; ++i) {
        sKn[tid * KS + 4 * i + 0] = qv[i].x * inv;
        sKn[tid * KS + 4 * i + 1] = qv[i].y * inv;
        sKn[tid * KS + 4 * i + 2] = qv[i].z * inv;
        sKn[tid * KS + 4 * i + 3] = qv[i].w * inv;
    }
    __syncthreads();

    sim_wmma(sKn, sMn, sSim, tid);
    __syncthreads();

    int best = 0; float bestv = -3.0e38f;
    for (int cl = 0; cl < CC; ++cl) {
        const float d0 = sSim[tid * KS + cl];
        if (d0 > bestv) { bestv = d0; best = cl; }     // first-max like argmax
    }
    atomicAdd(&sBins[best], 1);
#pragma unroll 4
    for (int i = 0; i < DD; ++i)
        atomicAdd(&sSums[best * DD + i], sKn[tid * KS + i]);
    __syncthreads();

    for (int i = tid; i < CC * DD; i += 256)
        atomicAdd(&g_sums[(long)h_idx * CC * DD + i], sSums[i]);
    if (tid < CC) atomicAdd(&g_bins[h_idx * CC + tid], sBins[tid]);
}

// ------------------------------------------------------------------
// K2: means_upd = bins==0 ? means : l2norm(sums)
// ------------------------------------------------------------------
__global__ __launch_bounds__(64) void kmr_update(const float* __restrict__ g_sums,
                                                 const int* __restrict__ g_bins,
                                                 const float* __restrict__ means,
                                                 float* __restrict__ means_upd) {
    __shared__ float red[64];
    const int hc = blockIdx.x;
    const int tid = threadIdx.x;
    const float s = g_sums[(long)hc * DD + tid];
    red[tid] = s * s;
    __syncthreads();
    for (int off = 32; off > 0; off >>= 1) {
        if (tid < off) red[tid] += red[tid + off];
        __syncthreads();
    }
    const float n = fmaxf(sqrtf(red[0]), 1e-12f);
    const float nv = s / n;
    const float old = means[(long)hc * DD + tid];
    means_upd[(long)hc * DD + tid] = (g_bins[hc] == 0) ? old : nv;
}

// ------------------------------------------------------------------
// K3: dists[b,h,c,t] = knorm(qk) . means_upd   (f32 WMMA)
// ------------------------------------------------------------------
__global__ __launch_bounds__(256) void kmr_dists(const float* __restrict__ qk,
                                                 const float* __restrict__ means_upd,
                                                 float* __restrict__ dists) {
    extern __shared__ char smD[];
    float* sKn  = (float*)smD;                  // 256 x KS
    float* sMn  = sKn + 256 * KS;               // 64 x KS
    float* sSim = sMn + 64 * KS;                // 256 x KS

    const int tid = threadIdx.x;
    const long gid0 = (long)blockIdx.x * 256;
    const long bh = gid0 / TT;
    const int h_idx = (int)(bh % HH);
    const int t0 = (int)(gid0 % TT);

    for (int i = tid; i < CC * DD; i += 256)
        sMn[(i >> 6) * KS + (i & 63)] = means_upd[(long)h_idx * CC * DD + i];

    const long gid = gid0 + tid;
    const float4* qp = (const float4*)(qk + gid * DD);
    float4 qv[16];
    float ss = 0.f;
#pragma unroll
    for (int i = 0; i < 16; ++i) {
        qv[i] = qp[i];
        ss += qv[i].x * qv[i].x + qv[i].y * qv[i].y + qv[i].z * qv[i].z + qv[i].w * qv[i].w;
    }
    const float inv = 1.0f / fmaxf(sqrtf(ss), 1e-12f);
#pragma unroll
    for (int i = 0; i < 16; ++i) {
        sKn[tid * KS + 4 * i + 0] = qv[i].x * inv;
        sKn[tid * KS + 4 * i + 1] = qv[i].y * inv;
        sKn[tid * KS + 4 * i + 2] = qv[i].z * inv;
        sKn[tid * KS + 4 * i + 3] = qv[i].w * inv;
    }
    __syncthreads();

    sim_wmma(sKn, sMn, sSim, tid);
    __syncthreads();

    // coalesced writeback: consecutive threads -> consecutive t
    for (int i = tid; i < CC * 256; i += 256) {
        const int cl = i >> 8;
        const int tc = i & 255;
        dists[(bh * CC + cl) * (long)TT + t0 + tc] = sSim[tc * KS + cl];
    }
}

// ------------------------------------------------------------------
// K4: per (b,h,c) top-128 (desc dist, idx tiebreak) + ascending idx sort
// ------------------------------------------------------------------
__global__ __launch_bounds__(256) void kmr_topk(const float* __restrict__ dists,
                                                int* __restrict__ indices) {
    extern __shared__ char sm4[];
    unsigned* ky = (unsigned*)sm4;
    int* ix = (int*)(sm4 + (size_t)TT * 4);
    int* sidx = (int*)(sm4 + (size_t)TT * 8);

    const int tid = threadIdx.x;
    const long bhc = blockIdx.x;
    const float* drow = dists + bhc * (long)TT;

    for (int i = tid; i < TT; i += 256) {
        unsigned u = __float_as_uint(drow[i]);
        u = (u & 0x80000000u) ? ~u : (u | 0x80000000u);
        ky[i] = u; ix[i] = i;
    }
    __syncthreads();

    for (unsigned k = 2; k <= (unsigned)TT; k <<= 1) {
        for (unsigned j = k >> 1; j > 0; j >>= 1) {
            for (int i = tid; i < TT; i += 256) {
                const unsigned p = (unsigned)i ^ j;
                if (p > (unsigned)i) {
                    const unsigned ka = ky[i], kb = ky[p];
                    const int ia = ix[i], ib = ix[p];
                    const bool before_pb = (kb > ka) || (kb == ka && ib < ia);
                    const bool descseg = ((i & k) == 0);
                    if (before_pb == descseg) {
                        ky[i] = kb; ky[p] = ka; ix[i] = ib; ix[p] = ia;
                    }
                }
            }
            __syncthreads();
        }
    }
    if (tid < WSZ) sidx[tid] = ix[tid];
    __syncthreads();

    for (unsigned k = 2; k <= (unsigned)WSZ; k <<= 1) {
        for (unsigned j = k >> 1; j > 0; j >>= 1) {
            for (int i = tid; i < WSZ; i += 256) {
                const unsigned p = (unsigned)i ^ j;
                if (p > (unsigned)i) {
                    const int a = sidx[i], bb2 = sidx[p];
                    const bool asc = ((i & k) == 0);
                    if ((bb2 < a) == asc) { sidx[i] = bb2; sidx[p] = a; }
                }
            }
            __syncthreads();
        }
    }
    if (tid < WSZ) indices[bhc * WSZ + tid] = sidx[tid];
}

// ------------------------------------------------------------------
// K5: windowed attention: TDM load of rel tile, bf16 WMMA, scatter-add
// ------------------------------------------------------------------
#define QS 72     // ushort stride for Q/K/R rows (64 + pad)
#define SS 132    // float stride for S/P rows (128 + pad)
#define AS 136    // ushort stride for attn / Vt rows (128 + pad)
#define SM_RF_OFF (128*QS*2*3 + 64*AS*2 + 128*SS*4*2 + 128*AS*2 + 128*4)

__global__ __launch_bounds__(256) void kmr_attn(const float* __restrict__ qk,
                                                const float* __restrict__ v,
                                                const float* __restrict__ relw,
                                                const int* __restrict__ indices,
                                                float* __restrict__ out,
                                                float* __restrict__ counts) {
    extern __shared__ char sm[];
    unsigned short* sQ  = (unsigned short*)sm;            // 128 x QS
    unsigned short* sK  = sQ + 128 * QS;                  // 128 x QS
    unsigned short* sR  = sK + 128 * QS;                  // 128 x QS (bf16)
    unsigned short* sVt = sR + 128 * QS;                  // 64  x AS (V^T)
    float* sS = (float*)(sVt + 64 * AS);                  // 128 x SS
    float* sP = sS + 128 * SS;                            // 128 x SS
    unsigned short* sA = (unsigned short*)(sP + 128 * SS);// 128 x AS
    int* sIdx = (int*)(sA + 128 * AS);                    // 128
    float* sRf = (float*)(sm + SM_RF_OFF);                // 128 x 64 f32 (TDM dest)

    const int tid = threadIdx.x;
    const int lane = tid & 31;
    const int wv = tid >> 5;
    const int half = lane >> 4;
    const int l15 = lane & 15;

    const int blk = blockIdx.x;
    const long bh = blk / NCW;
    const int w = blk % NCW;
    const int h_idx = (int)(bh % HH);
    const float scale = 0.125f;

    // ---- TDM: async 2D tile load of rel_weights[:, h, :] into LDS ----
    // tensor: x = 64 f32 (contiguous), y = 128 rows, y-stride = H*D elements
    if (tid < 32) {
        const unsigned lds_off = (unsigned)(unsigned long long)(void*)sRf;
        const unsigned long long ga =
            (unsigned long long)(const void*)(relw + (long)h_idx * DD);
        u32x4 g0;
        g0[0] = 1u;                                           // count=1 (valid D#)
        g0[1] = lds_off;                                      // lds_addr
        g0[2] = (unsigned)(ga & 0xFFFFFFFFu);                 // global_addr[31:0]
        g0[3] = (unsigned)((ga >> 32) & 0x01FFFFFFu) | (2u << 30); // addr[56:32], type=2
        i32x8 g1;
        g1[0] = (int)(2u << 16);                              // data_size = 4 bytes
        g1[1] = (int)(64u << 16);                             // tensor_dim0 = 64
        g1[2] = (int)(128u << 16);                            // tensor_dim1 = 128
        g1[3] = (int)(64u << 16);                             // tile_dim0 = 64
        g1[4] = (int)128;                                     // tile_dim1 = 128
        g1[5] = (int)(HH * DD);                               // dim0_stride = 1024
        g1[6] = 0; g1[7] = 0;
        i32x4 gz = {0, 0, 0, 0};
        i32x8 gz8 = {0, 0, 0, 0, 0, 0, 0, 0};
        __builtin_amdgcn_tensor_load_to_lds(g0, g1, gz, gz, gz8, 0);
    }

    // ---- gather Q/K/V (overlaps the TDM transfer) ----
    if (tid < WSZ) {
        const int tok = indices[bh * (long)TT + (long)w * WSZ + tid];
        sIdx[tid] = tok;
        const float4* qp = (const float4*)(qk + (bh * (long)TT + tok) * DD);
        const float4* vp = (const float4*)(v + (bh * (long)TT + tok) * DD);
        __builtin_prefetch(vp, 0, 1);
        float4 q4[16]; float ss = 0.f;
#pragma unroll
        for (int i = 0; i < 16; ++i) {
            q4[i] = qp[i];
            ss += q4[i].x * q4[i].x + q4[i].y * q4[i].y + q4[i].z * q4[i].z + q4[i].w * q4[i].w;
        }
        const float inv = 1.f / fmaxf(sqrtf(ss), 1e-12f);
#pragma unroll
        for (int i = 0; i < 16; ++i) {
            sQ[tid * QS + 4 * i + 0] = f2bf(q4[i].x);
            sQ[tid * QS + 4 * i + 1] = f2bf(q4[i].y);
            sQ[tid * QS + 4 * i + 2] = f2bf(q4[i].z);
            sQ[tid * QS + 4 * i + 3] = f2bf(q4[i].w);
            sK[tid * QS + 4 * i + 0] = f2bf(q4[i].x * inv);
            sK[tid * QS + 4 * i + 1] = f2bf(q4[i].y * inv);
            sK[tid * QS + 4 * i + 2] = f2bf(q4[i].z * inv);
            sK[tid * QS + 4 * i + 3] = f2bf(q4[i].w * inv);
        }
#pragma unroll
        for (int i = 0; i < 16; ++i) {
            float4 vv = vp[i];
            sVt[(4 * i + 0) * AS + tid] = f2bf(vv.x);
            sVt[(4 * i + 1) * AS + tid] = f2bf(vv.y);
            sVt[(4 * i + 2) * AS + tid] = f2bf(vv.z);
            sVt[(4 * i + 3) * AS + tid] = f2bf(vv.w);
        }
    }

    // wave 0 drains TENSORcnt; barrier publishes sRf to all waves
    if (tid < 32) __builtin_amdgcn_s_wait_tensorcnt(0);
    __syncthreads();

    // all threads convert rel tile f32 -> bf16
    for (int i = tid; i < WSZ * DD; i += 256)
        sR[(i >> 6) * QS + (i & 63)] = f2bf(sRf[i]);
    __syncthreads();

    // ---- S = Q*K^T, P = Q*R^T ----
    for (int nt = 0; nt < 8; ++nt) {
        v8f acc = {0.f, 0.f, 0.f, 0.f, 0.f, 0.f, 0.f, 0.f};
        v8f accp = {0.f, 0.f, 0.f, 0.f, 0.f, 0.f, 0.f, 0.f};
#pragma unroll
        for (int kb = 0; kb < 2; ++kb) {
            v16bf a  = frag_a(sQ + (wv * 16 + l15) * QS, kb, half);
            v16bf bk = frag_b(sK + (nt * 16 + l15) * QS, kb, half);
            v16bf br = frag_b(sR + (nt * 16 + l15) * QS, kb, half);
            acc  = __builtin_amdgcn_wmma_f32_16x16x32_bf16(false, a, false, bk,
                                                           (short)0, acc, false, false);
            accp = __builtin_amdgcn_wmma_f32_16x16x32_bf16(false, a, false, br,
                                                           (short)0, accp, false, false);
        }
#pragma unroll
        for (int r = 0; r < 8; ++r) {
            const int mm = wv * 16 + half * 8 + r;
            sS[mm * SS + nt * 16 + l15] = acc[r];
            sP[mm * SS + nt * 16 + l15] = accp[r];
        }
    }
    __syncthreads();

    // ---- shift-bias + diag mask + f32 softmax, A -> bf16 ----
    if (tid < WSZ) {
        const int i = tid;
        float mx = -3.0e38f;
        for (int j = 0; j < WSZ; ++j) {
            float l = sS[i * SS + j] * scale;
            if (j <= i) l += sP[i * SS + (WSZ - 1 + j - i)] * scale;
            if (j == i) l = -50000.f;
            sS[i * SS + j] = l;
            mx = fmaxf(mx, l);
        }
        float sum = 0.f;
        for (int j = 0; j < WSZ; ++j) {
            const float e = __expf(sS[i * SS + j] - mx);
            sS[i * SS + j] = e;
            sum += e;
        }
        const float rinv = 1.f / sum;
        for (int j = 0; j < WSZ; ++j) sA[i * AS + j] = f2bf(sS[i * SS + j] * rinv);
    }
    __syncthreads();

    // ---- O = A * V, scatter-add ----
    for (int nt = 0; nt < 4; ++nt) {
        v8f acc = {0.f, 0.f, 0.f, 0.f, 0.f, 0.f, 0.f, 0.f};
#pragma unroll
        for (int kb = 0; kb < 4; ++kb) {
            v16bf a  = frag_a(sA + (wv * 16 + l15) * AS, kb, half);
            v16bf bv = frag_b(sVt + (nt * 16 + l15) * AS, kb, half);
            acc = __builtin_amdgcn_wmma_f32_16x16x32_bf16(false, a, false, bv,
                                                          (short)0, acc, false, false);
        }
#pragma unroll
        for (int r = 0; r < 8; ++r) {
            const int mm = wv * 16 + half * 8 + r;
            const int tok = sIdx[mm];
            atomicAdd(out + (bh * (long)TT + tok) * DD + nt * 16 + l15, acc[r]);
        }
    }
    if (tid < WSZ) atomicAdd(counts + bh * (long)TT + sIdx[tid], 1.0f);
}

// ------------------------------------------------------------------
// K6: out /= (counts + 1e-5)
// ------------------------------------------------------------------
__global__ __launch_bounds__(256) void kmr_norm(float* __restrict__ out,
                                                const float* __restrict__ counts) {
    const long gid = (long)blockIdx.x * 256 + threadIdx.x;
    out[gid] = out[gid] / (counts[gid >> 6] + 1e-5f);
}

// ------------------------------------------------------------------
extern "C" void kernel_launch(void* const* d_in, const int* in_sizes, int n_in,
                              void* d_out, int out_size, void* d_ws, size_t ws_size,
                              hipStream_t stream) {
    (void)in_sizes; (void)n_in; (void)ws_size;
    const float* qk    = (const float*)d_in[0];
    const float* v     = (const float*)d_in[1];
    const float* means = (const float*)d_in[2];
    const float* relw  = (const float*)d_in[3];
    float* out = (float*)d_out;
    char* ws = (char*)d_ws;

    size_t o = 0;
    float* g_sums    = (float*)(ws + o); o += (size_t)HH * CC * DD * 4;
    int*   g_bins    = (int*)(ws + o);   o += 4096;
    float* means_upd = (float*)(ws + o); o += (size_t)HH * CC * DD * 4;
    float* counts    = (float*)(ws + o); o += (size_t)BB * HH * TT * 4;
    int*   indices   = (int*)(ws + o);   o += (size_t)BB * HH * TT * 4;
    float* dists     = (float*)(ws + o); o += (size_t)BB * HH * CC * TT * 4;

    (void)hipMemsetAsync(g_sums, 0, (size_t)HH * CC * DD * 4, stream);
    (void)hipMemsetAsync(g_bins, 0, (size_t)HH * CC * 4, stream);
    (void)hipMemsetAsync(counts, 0, (size_t)BB * HH * TT * 4, stream);
    (void)hipMemsetAsync(out, 0, (size_t)out_size * 4, stream);

    const int ntok_blocks = (BB * HH * TT) / 256;   // 2048

    const size_t smem_assign = ((size_t)256 * KS + 64 * KS + 256 * KS + 64 * 64) * 4 + 256;
    kmr_assign<<<ntok_blocks, 256, smem_assign, stream>>>(qk, means, g_sums, g_bins);

    kmr_update<<<HH * CC, 64, 0, stream>>>(g_sums, g_bins, means, means_upd);

    const size_t smem_dists = ((size_t)256 * KS + 64 * KS + 256 * KS) * 4;
    kmr_dists<<<ntok_blocks, 256, smem_dists, stream>>>(qk, means_upd, dists);

    const size_t smem_topk = (size_t)TT * 8 + WSZ * 4;
    kmr_topk<<<BB * HH * CC, 256, smem_topk, stream>>>(dists, indices);

    const size_t smem_attn = (size_t)SM_RF_OFF + (size_t)WSZ * DD * 4;
    kmr_attn<<<BB * HH * NCW, 256, smem_attn, stream>>>(qk, v, relw, indices, out, counts);

    kmr_norm<<<(int)(((size_t)BB * HH * TT * DD) / 256), 256, 0, stream>>>(out, counts);
}
